// SNN_28613072126563
// MI455X (gfx1250) — matrix operand order repaired and checked by reading, not compiled, not run on hardware
//
#include <hip/hip_runtime.h>
#include <hip/hip_bf16.h>

typedef __attribute__((ext_vector_type(16))) __bf16 v16bf;
typedef __attribute__((ext_vector_type(8)))  __bf16 v8bf;
typedef __attribute__((ext_vector_type(8)))  float  v8f;

#define ALPHA_MIN_C 0.81873075307798185867f  /* exp(-1/5)  */
#define ALPHA_MAX_C 0.96078943915232320784f  /* exp(-1/25) */

// ---------------------------------------------------------------------------
// Generic bf16 WMMA GEMM:  C[M,N] = A[M,K] * Bn[N,K]^T   (Bn rows are output
// columns, K-contiguous -> each B-fragment lane is one 32B contiguous load).
// Block = 256 threads = 8 waves, arranged WM x WN waves of 16x16 tiles.
// ---------------------------------------------------------------------------
template <int WN>
__global__ __launch_bounds__(256) void gemm_bf16_wmma(
    const __bf16* __restrict__ A, const __bf16* __restrict__ Bn,
    float* __restrict__ C, int K, int lda, int ldb, int ldc) {
  constexpr int WM = 8 / WN;
  const int w    = threadIdx.x >> 5;
  const int lane = threadIdx.x & 31;
  const int half = lane >> 4;     // which 16-lane half of the wave
  const int l16  = lane & 15;
  const int m0 = blockIdx.y * (WM * 16) + (w / WN) * 16;
  const int n0 = blockIdx.x * (WN * 16) + (w % WN) * 16;

  // A 16x32 bf16 fragment: lane half 0 -> K 0..15, half 1 -> K 16..31 (contiguous)
  const __bf16* arow = A  + (size_t)(m0 + l16) * lda + half * 16;
  // B 32x16 bf16 fragment: lane half 0 -> K 0..15, half 1 -> K 16..31 (contiguous)
  const __bf16* brow = Bn + (size_t)(n0 + l16) * ldb + half * 16;

  v8f c = {};
#pragma unroll 4
  for (int k0 = 0; k0 < K; k0 += 32) {
    union { v16bf v; v8bf h[2]; } a;
    a.h[0] = *(const v8bf*)(arow + k0);
    a.h[1] = *(const v8bf*)(arow + k0 + 8);
    v16bf b = *(const v16bf*)(brow + k0);
    c = __builtin_amdgcn_wmma_f32_16x16x32_bf16(false, a.v, false, b,
                                                (short)0, c, false, false);
  }
  float* crow = C + (size_t)(m0 + half * 8) * ldc + n0 + l16;
#pragma unroll
  for (int v = 0; v < 8; ++v) crow[(size_t)v * ldc] = c[v];
}

// ---------------------------------------------------------------------------
// One recurrent RLIF time step, fused:  r = st_in * Vz^T (WMMA, K=1024),
// then ut = a*(ut-st) + (1-a)*(Wx[:,t,:] + r);  st_out = (ut > 1).
// Grid covers the [B=128, H=1024] state; 32x64 block tile, 8 waves.
// ---------------------------------------------------------------------------
__global__ __launch_bounds__(256) void rlif_step_kernel(
    const __bf16* __restrict__ stin,  // [128,1024] bf16 spikes (A matrix)
    const __bf16* __restrict__ VT,    // [1024,1024] bf16, VT[n][k]=V[k][n], diag 0
    const float*  __restrict__ Wx,    // [B*T,1024] f32, row = b*T + t
    const float*  __restrict__ alpha, // [1024] clipped
    float*        __restrict__ ut,    // [128,1024] f32 state (RMW)
    __bf16*       __restrict__ stout, // [128,1024] bf16 next spikes
    __bf16*       __restrict__ sstore,// [B*T,1024] bf16 spike history
    int t, int T) {
  const int H = 1024;
  const int w = threadIdx.x >> 5, lane = threadIdx.x & 31;
  const int half = lane >> 4, l16 = lane & 15;
  const int m0 = blockIdx.y * 32 + (w >> 2) * 16;   // batch tile
  const int n0 = blockIdx.x * 64 + (w & 3) * 16;    // hidden tile

  const __bf16* arow = stin + (size_t)(m0 + l16) * H + half * 16;
  const __bf16* brow = VT   + (size_t)(n0 + l16) * H + half * 16;

  v8f c = {};
#pragma unroll 8
  for (int k0 = 0; k0 < H; k0 += 32) {
    union { v16bf v; v8bf h[2]; } a;
    a.h[0] = *(const v8bf*)(arow + k0);
    a.h[1] = *(const v8bf*)(arow + k0 + 8);
    v16bf b = *(const v16bf*)(brow + k0);
    c = __builtin_amdgcn_wmma_f32_16x16x32_bf16(false, a.v, false, b,
                                                (short)0, c, false, false);
  }

  const int h = n0 + l16;
  const float a = alpha[h];
#pragma unroll
  for (int v = 0; v < 8; ++v) {
    const int m = m0 + half * 8 + v;                 // batch index
    const size_t si = (size_t)m * H + h;
    float u     = ut[si];
    float sprev = (float)stin[si];
    float wx    = Wx[((size_t)m * T + t) * H + h];
    u = a * (u - sprev) + (1.0f - a) * (wx + c[v]);
    float sn = (u > 1.0f) ? 1.0f : 0.0f;
    ut[si] = u;
    __bf16 sb = (__bf16)sn;
    stout[si] = sb;
    sstore[((size_t)m * T + t) * H + h] = sb;
  }
}

// ---------------------------------------------------------------------------
// Prep / pointwise kernels
// ---------------------------------------------------------------------------
__global__ void k_cast_pad(const float* __restrict__ src, __bf16* __restrict__ dst,
                           int srows, int scols, int drows, int dcols) {
  size_t i = (size_t)blockIdx.x * blockDim.x + threadIdx.x;
  size_t n = (size_t)drows * dcols;
  if (i >= n) return;
  int r = (int)(i / dcols), c = (int)(i % dcols);
  float v = (r < srows && c < scols) ? src[(size_t)r * scols + c] : 0.0f;
  dst[i] = (__bf16)v;
}

__global__ void k_transpose_zd(const float* __restrict__ V, __bf16* __restrict__ VT, int n) {
  size_t i = (size_t)blockIdx.x * blockDim.x + threadIdx.x;
  if (i >= (size_t)n * n) return;
  int r = (int)(i / n), c = (int)(i % n);          // VT[r][c] = V[c][r]
  VT[i] = (r == c) ? (__bf16)0.0f : (__bf16)V[(size_t)c * n + r];
}

__global__ void k_clip(const float* __restrict__ a, float* __restrict__ o, int n) {
  int i = blockIdx.x * blockDim.x + threadIdx.x;
  if (i < n) o[i] = fminf(fmaxf(a[i], ALPHA_MIN_C), ALPHA_MAX_C);
}

// ---------------------------------------------------------------------------
// Readout: sequential LIF + cumulative softmax. One wave per batch element.
// wxo is [B*T, 32] f32 (cols 20..31 are exact zeros since Wout pad rows are 0).
// ---------------------------------------------------------------------------
__global__ __launch_bounds__(32) void readout_scan_kernel(
    const float* __restrict__ wxo, const float* __restrict__ ao,
    const float* __restrict__ ut0, float* __restrict__ out, int T) {
  const int b = blockIdx.x, o = threadIdx.x;
  const bool act = o < 20;
  float a = act ? ao[o] : 0.0f;
  float u = act ? ut0[b * 20 + o] : 0.0f;
  float acc = 0.0f;
  const float* row = wxo + (size_t)b * T * 32 + o;
  for (int t = 0; t < T; ++t) {
    float wx = act ? row[(size_t)t * 32] : 0.0f;
    u = a * u + (1.0f - a) * wx;
    float val = act ? u : -3.0e38f;
    float mx = val;
    for (int off = 16; off > 0; off >>= 1) mx = fmaxf(mx, __shfl_xor(mx, off, 32));
    float e = act ? __expf(u - mx) : 0.0f;
    float s = e;
    for (int off = 16; off > 0; off >>= 1) s += __shfl_xor(s, off, 32);
    acc += e / s;
  }
  if (act) out[b * 20 + o] = acc;
}

// ---------------------------------------------------------------------------
extern "C" void kernel_launch(void* const* d_in, const int* in_sizes, int n_in,
                              void* d_out, int out_size, void* d_ws, size_t ws_size,
                              hipStream_t stream) {
  (void)in_sizes; (void)n_in; (void)out_size; (void)ws_size;
  const int Bb = 128, T = 500, F = 700, Fp = 704, H = 1024, NO = 20;
  const int BT = Bb * T;

  const float* x      = (const float*)d_in[0];
  const float* W1     = (const float*)d_in[1];
  const float* V1     = (const float*)d_in[2];
  const float* alpha1 = (const float*)d_in[3];
  const float* W2     = (const float*)d_in[4];
  const float* V2     = (const float*)d_in[5];
  const float* alpha2 = (const float*)d_in[6];
  const float* Wout   = (const float*)d_in[7];
  const float* alphao = (const float*)d_in[8];
  const float* ut0_1  = (const float*)d_in[9];
  const float* st0_1  = (const float*)d_in[10];
  const float* ut0_2  = (const float*)d_in[11];
  const float* st0_2  = (const float*)d_in[12];
  const float* ut0_r  = (const float*)d_in[13];

  char* wsp = (char*)d_ws;
  auto alloc = [&](size_t bytes) -> void* {
    void* p = (void*)wsp;
    wsp += (bytes + 255) & ~(size_t)255;
    return p;
  };
  __bf16* xb    = (__bf16*)alloc((size_t)BT * Fp * 2);
  __bf16* W1b   = (__bf16*)alloc((size_t)H * Fp * 2);
  __bf16* W2b   = (__bf16*)alloc((size_t)H * H * 2);
  __bf16* V1T   = (__bf16*)alloc((size_t)H * H * 2);
  __bf16* V2T   = (__bf16*)alloc((size_t)H * H * 2);
  __bf16* WoutB = (__bf16*)alloc((size_t)32 * H * 2);
  float*  Wx1   = (float*) alloc((size_t)BT * H * 4);
  float*  Wx2   = (float*) alloc((size_t)BT * H * 4);
  __bf16* s1b   = (__bf16*)alloc((size_t)BT * H * 2);
  __bf16* s2b   = (__bf16*)alloc((size_t)BT * H * 2);
  float*  ut1   = (float*) alloc((size_t)Bb * H * 4);
  float*  ut2   = (float*) alloc((size_t)Bb * H * 4);
  __bf16* stA1  = (__bf16*)alloc((size_t)Bb * H * 2);
  __bf16* stB1  = (__bf16*)alloc((size_t)Bb * H * 2);
  __bf16* stA2  = (__bf16*)alloc((size_t)Bb * H * 2);
  __bf16* stB2  = (__bf16*)alloc((size_t)Bb * H * 2);
  float*  a1c   = (float*) alloc((size_t)H * 4);
  float*  a2c   = (float*) alloc((size_t)H * 4);
  float*  aoc   = (float*) alloc((size_t)NO * 4);
  float*  wxo   = (float*) alloc((size_t)BT * 32 * 4);

  auto g1 = [](size_t n) { return dim3((unsigned)((n + 255) / 256)); };

  // ---- prep: casts, padding, transposes, clips, state init ----
  k_cast_pad<<<g1((size_t)BT * Fp), 256, 0, stream>>>(x, xb, BT, F, BT, Fp);
  k_cast_pad<<<g1((size_t)H * Fp), 256, 0, stream>>>(W1, W1b, H, F, H, Fp);
  k_cast_pad<<<g1((size_t)H * H), 256, 0, stream>>>(W2, W2b, H, H, H, H);
  k_transpose_zd<<<g1((size_t)H * H), 256, 0, stream>>>(V1, V1T, H);
  k_transpose_zd<<<g1((size_t)H * H), 256, 0, stream>>>(V2, V2T, H);
  k_cast_pad<<<g1((size_t)32 * H), 256, 0, stream>>>(Wout, WoutB, NO, H, 32, H);
  k_clip<<<g1(H), 256, 0, stream>>>(alpha1, a1c, H);
  k_clip<<<g1(H), 256, 0, stream>>>(alpha2, a2c, H);
  k_clip<<<g1(NO), 256, 0, stream>>>(alphao, aoc, NO);
  hipMemcpyAsync(ut1, ut0_1, (size_t)Bb * H * 4, hipMemcpyDeviceToDevice, stream);
  hipMemcpyAsync(ut2, ut0_2, (size_t)Bb * H * 4, hipMemcpyDeviceToDevice, stream);
  k_cast_pad<<<g1((size_t)Bb * H), 256, 0, stream>>>(st0_1, stA1, Bb, H, Bb, H);
  k_cast_pad<<<g1((size_t)Bb * H), 256, 0, stream>>>(st0_2, stA2, Bb, H, Bb, H);

  // ---- layer 1 feedforward GEMM: Wx1[BT,H] = xb[BT,Fp] * W1b[H,Fp]^T ----
  gemm_bf16_wmma<4><<<dim3(H / 64, BT / 32), 256, 0, stream>>>(
      xb, W1b, Wx1, Fp, Fp, Fp, H);

  // ---- layer 1 recurrent scan (500 fused WMMA steps) ----
  dim3 sgrid(H / 64, Bb / 32);  // (16,4)
  for (int t = 0; t < T; ++t) {
    const __bf16* si = (t & 1) ? stB1 : stA1;
    __bf16*       so = (t & 1) ? stA1 : stB1;
    rlif_step_kernel<<<sgrid, 256, 0, stream>>>(si, V1T, Wx1, a1c, ut1, so, s1b, t, T);
  }

  // ---- layer 2 feedforward GEMM: Wx2 = s1b * W2b^T ----
  gemm_bf16_wmma<4><<<dim3(H / 64, BT / 32), 256, 0, stream>>>(
      s1b, W2b, Wx2, H, H, H, H);

  // ---- layer 2 recurrent scan ----
  for (int t = 0; t < T; ++t) {
    const __bf16* si = (t & 1) ? stB2 : stA2;
    __bf16*       so = (t & 1) ? stA2 : stB2;
    rlif_step_kernel<<<sgrid, 256, 0, stream>>>(si, V2T, Wx2, a2c, ut2, so, s2b, t, T);
  }

  // ---- readout GEMM: wxo[BT,32] = s2b[BT,H] * WoutB[32,H]^T ----
  gemm_bf16_wmma<2><<<dim3(32 / 32, BT / 64), 256, 0, stream>>>(
      s2b, WoutB, wxo, H, H, H, 32);

  // ---- readout scan: LIF + cumulative softmax -> d_out [128,20] ----
  readout_scan_kernel<<<Bb, 32, 0, stream>>>(wxo, aoc, ut0_r, (float*)d_out, T);
}
